// JiTAttentionBlock_46325517255306
// MI455X (gfx1250) — compile-verified, matrix operand order checked by reading, not run on hardware
//
#include <hip/hip_runtime.h>
#include <hip/hip_bf16.h>

// ---------------------------------------------------------------------------
// Types for CDNA5 WMMA (wave32, V_WMMA_F32_16X16X32_BF16)
// ---------------------------------------------------------------------------
typedef __attribute__((ext_vector_type(16))) __bf16 v16bf;
typedef __attribute__((ext_vector_type(8)))  __bf16 v8bf;
typedef __attribute__((ext_vector_type(8)))  float  v8f;

#define EPSF 1e-6f

// Load a 16-half fragment in the CDNA5 16-bit A/B VGPR layout:
// lane holds 8 halves at p[0..7] and 8 halves at p[16..23].
__device__ __forceinline__ v16bf load_frag(const __bf16* __restrict__ p) {
    v8bf lo = *reinterpret_cast<const v8bf*>(p);
    v8bf hi = *reinterpret_cast<const v8bf*>(p + 16);
    v16bf r;
#pragma unroll
    for (int i = 0; i < 8; ++i) { r[i] = lo[i]; r[i + 8] = hi[i]; }
    return r;
}

__device__ __forceinline__ v8f wmma_bf16(v16bf a, v16bf b, v8f c) {
    return __builtin_amdgcn_wmma_f32_16x16x32_bf16(false, a, false, b,
                                                   (short)0, c, false, false);
}

// ---------------------------------------------------------------------------
// f32 -> bf16 elementwise convert (weights)
// ---------------------------------------------------------------------------
__global__ __launch_bounds__(256) void k_tobf16(const float* __restrict__ in,
                                                __bf16* __restrict__ out, int n) {
    int i = blockIdx.x * 256 + threadIdx.x;
    if (i < n) out[i] = (__bf16)in[i];
}

// ---------------------------------------------------------------------------
// adaLN: ada = silu(temb) @ ada_w.T + ada_b     (16 x 3072, K=512)
// ---------------------------------------------------------------------------
__global__ __launch_bounds__(256) void k_adaln(const float* __restrict__ temb,
                                               const float* __restrict__ W,
                                               const float* __restrict__ bias,
                                               float* __restrict__ ada) {
    __shared__ float st[512];
    int b = blockIdx.x, t = threadIdx.x;
    for (int i = t; i < 512; i += 256) {
        float v = temb[b * 512 + i];
        st[i] = v / (1.f + __expf(-v));
    }
    __syncthreads();
    for (int o = t; o < 3072; o += 256) {
        const float* wr = W + (size_t)o * 512;
        float acc = 0.f;
        for (int kx = 0; kx < 512; ++kx) acc += st[kx] * wr[kx];
        ada[b * 3072 + o] = acc + bias[o];
    }
}

// ---------------------------------------------------------------------------
// GroupNorm(32) over x (B,C,H,W) -> h (B,N,C) f32 ; one block per (b,g)
// ---------------------------------------------------------------------------
__global__ __launch_bounds__(256) void k_gnorm(const float* __restrict__ x,
                                               const float* __restrict__ gw,
                                               const float* __restrict__ gb,
                                               float* __restrict__ h) {
    __shared__ float rs[256], rq[256];
    int b = blockIdx.x >> 5, g = blockIdx.x & 31;
    const float* base = x + ((size_t)b * 512 + g * 16) * 1024;
    int t = threadIdx.x;
    float s = 0.f, q = 0.f;
    for (int i = t; i < 16384; i += 256) { float v = base[i]; s += v; q += v * v; }
    rs[t] = s; rq[t] = q; __syncthreads();
    for (int off = 128; off; off >>= 1) {
        if (t < off) { rs[t] += rs[t + off]; rq[t] += rq[t + off]; }
        __syncthreads();
    }
    float mu = rs[0] * (1.f / 16384.f);
    float var = rq[0] * (1.f / 16384.f) - mu * mu;
    float inv = rsqrtf(var + EPSF);
    for (int i = t; i < 16384; i += 256) {
        int cc = i >> 10, n = i & 1023, c = g * 16 + cc;
        float v = (base[i] - mu) * inv * gw[c] + gb[c];
        h[((size_t)b * 1024 + n) * 512 + c] = v;
    }
}

// ---------------------------------------------------------------------------
// RMSNorm + adaLN modulate -> bf16 ; one block per token
// ---------------------------------------------------------------------------
__global__ __launch_bounds__(256) void k_rmsmod(const float* __restrict__ h,
                                                const float* __restrict__ w,
                                                const float* __restrict__ ada,
                                                int shiftOff, int scaleOff,
                                                __bf16* __restrict__ out) {
    __shared__ float red[256];
    int m = blockIdx.x, b = m >> 10, t = threadIdx.x;
    const float* row = h + (size_t)m * 512;
    float a0 = row[t], a1 = row[t + 256];
    red[t] = a0 * a0 + a1 * a1;
    __syncthreads();
    for (int off = 128; off; off >>= 1) {
        if (t < off) red[t] += red[t + off];
        __syncthreads();
    }
    float inv = rsqrtf(red[0] * (1.f / 512.f) + EPSF);
    const float* ab = ada + b * 3072;
    for (int c = t; c < 512; c += 256) {
        float v = row[c] * inv * w[c];
        v = v * (1.f + ab[scaleOff + c]) + ab[shiftOff + c];
        out[(size_t)m * 512 + c] = (__bf16)v;
    }
}

// ---------------------------------------------------------------------------
// Generic WMMA GEMM: C[m,n] = sum_k A[m,k] * W[n,k] (+bias)
// A bf16 (M x K row-major), W bf16 (N x K row-major). Wave tile 64x64.
// mode 0: outF[m,n] = acc + bias[n]
// mode 1: hRes[(mGlob+m)*512 + n] += ada[b*3072+gateOff+n] * (acc + bias[n])
// ---------------------------------------------------------------------------
__global__ __launch_bounds__(32) void k_gemm(const __bf16* __restrict__ A,
                                             const __bf16* __restrict__ W,
                                             const float* __restrict__ bias,
                                             float* __restrict__ outF,
                                             float* __restrict__ hRes,
                                             const float* __restrict__ ada,
                                             int K, int Nout, int nTiles,
                                             int mGlobalOff, int gateOff, int mode) {
    int bid = blockIdx.x;
    int m0 = (bid / nTiles) * 64;
    int n0 = (bid % nTiles) * 64;
    int lane = threadIdx.x;
    int r15 = lane & 15;
    int half8 = (lane >> 4) * 8;

    v8f acc[4][4] = {};
    for (int kk = 0; kk < 32 * (K / 32); kk += 32) {
        __builtin_prefetch(A + (size_t)(m0 + r15) * K + kk + 64, 0, 1);
        v16bf a[4];
#pragma unroll
        for (int mi = 0; mi < 4; ++mi)
            a[mi] = load_frag(A + (size_t)(m0 + mi * 16 + r15) * K + kk + half8);
#pragma unroll
        for (int ni = 0; ni < 4; ++ni) {
            v16bf bfrag = load_frag(W + (size_t)(n0 + ni * 16 + r15) * K + kk + half8);
#pragma unroll
            for (int mi = 0; mi < 4; ++mi)
                acc[mi][ni] = wmma_bf16(a[mi], bfrag, acc[mi][ni]);
        }
    }

    int rbase = (lane >> 4) * 8;  // rows 0..7 (lanes 0-15) or 8..15 (lanes 16-31)
#pragma unroll
    for (int mi = 0; mi < 4; ++mi) {
#pragma unroll
        for (int ni = 0; ni < 4; ++ni) {
            int col = n0 + ni * 16 + r15;
            float bv = bias[col];
#pragma unroll
            for (int r = 0; r < 8; ++r) {
                int row = m0 + mi * 16 + rbase + r;
                float v = acc[mi][ni][r] + bv;
                if (mode == 0) {
                    outF[(size_t)row * Nout + col] = v;
                } else {
                    int gm = mGlobalOff + row;
                    float g = ada[(gm >> 10) * 3072 + gateOff + col];
                    hRes[(size_t)gm * 512 + col] += g * v;
                }
            }
        }
    }
}

// ---------------------------------------------------------------------------
// q/k RMSNorm + 2D axial RoPE + layout. qkv f32 chunk (Mc x 1536).
// Writes q,k bf16 (BH,N,64) and vT bf16 (BH,64,N). One wave per head.
// ---------------------------------------------------------------------------
__global__ __launch_bounds__(256) void k_ropeqkv(const float* __restrict__ qkv,
                                                 const float* __restrict__ qnw,
                                                 const float* __restrict__ knw,
                                                 __bf16* __restrict__ qo,
                                                 __bf16* __restrict__ ko,
                                                 __bf16* __restrict__ vTo, int m0) {
    int mloc = blockIdx.x;
    int m = m0 + mloc;
    int b = m >> 10, n = m & 1023;
    int t = threadIdx.x;
    int h = t >> 5, ln = t & 31;
    int d0 = ln * 2, d1 = d0 + 1;
    const float* row = qkv + (size_t)mloc * 1536;
    float q0 = row[h * 64 + d0], q1 = row[h * 64 + d1];
    float k0 = row[512 + h * 64 + d0], k1 = row[512 + h * 64 + d1];
    float v0 = row[1024 + h * 64 + d0], v1 = row[1024 + h * 64 + d1];

    float sq = q0 * q0 + q1 * q1, sk = k0 * k0 + k1 * k1;
#pragma unroll
    for (int msk = 16; msk; msk >>= 1) {
        sq += __shfl_xor(sq, msk, 32);
        sk += __shfl_xor(sk, msk, 32);
    }
    float rq = rsqrtf(sq * (1.f / 64.f) + EPSF);
    float rk = rsqrtf(sk * (1.f / 64.f) + EPSF);
    q0 *= rq * qnw[d0]; q1 *= rq * qnw[d1];
    k0 *= rk * knw[d0]; k1 *= rk * knw[d1];

    // 2D axial RoPE: d<32 -> y axis, d>=32 -> x axis; freq = 10000^(-(e>>1)/16)
    float pos = (d0 < 32) ? (float)(n >> 5) : (float)(n & 31);
    int e = (d0 < 32) ? d0 : (d0 - 32);
    float fr = __powf(10000.f, -(float)(e >> 1) * (1.f / 16.f));
    float ang = pos * fr;
    float c = __cosf(ang), s = __sinf(ang);
    float Q0 = q0 * c - q1 * s, Q1 = q1 * c + q0 * s;
    float K0 = k0 * c - k1 * s, K1 = k1 * c + k0 * s;

    int bh = b * 8 + h;
    size_t qoff = (size_t)bh * 1024 * 64 + (size_t)n * 64;
    qo[qoff + d0] = (__bf16)Q0; qo[qoff + d1] = (__bf16)Q1;
    ko[qoff + d0] = (__bf16)K0; ko[qoff + d1] = (__bf16)K1;
    size_t voff = (size_t)bh * 64 * 1024;
    vTo[voff + (size_t)d0 * 1024 + n] = (__bf16)v0;
    vTo[voff + (size_t)d1 * 1024 + n] = (__bf16)v1;
}

// ---------------------------------------------------------------------------
// Flash attention: one wave per (bh, 16-row q tile). N=1024, hd=64.
// S = Q K^T via 4 WMMA per 32-key block; online softmax; P through LDS
// (C-layout -> A-layout transpose); O += P V via 4 WMMA. Writes o bf16 (B,N,C).
// ---------------------------------------------------------------------------
__global__ __launch_bounds__(32) void k_fattn(const __bf16* __restrict__ q,
                                              const __bf16* __restrict__ k,
                                              const __bf16* __restrict__ vT,
                                              __bf16* __restrict__ o) {
    __shared__ __bf16 pbuf[16 * 32];
    int bh = blockIdx.y, qt = blockIdx.x;
    int lane = threadIdx.x;
    int r15 = lane & 15;
    int halfsel = lane >> 4;
    int half8 = halfsel * 8;

    const __bf16* qb = q + (size_t)bh * 1024 * 64 + (size_t)qt * 16 * 64;
    v16bf qf0 = load_frag(qb + r15 * 64 + half8);
    v16bf qf1 = load_frag(qb + r15 * 64 + 32 + half8);

    const __bf16* kb = k + (size_t)bh * 1024 * 64;
    const __bf16* vb = vT + (size_t)bh * 64 * 1024;

    v8f o0 = {}, o1 = {}, o2 = {}, o3 = {};
    float mrow[8], lrow[8];
#pragma unroll
    for (int r = 0; r < 8; ++r) { mrow[r] = -1e30f; lrow[r] = 0.f; }

    for (int j = 0; j < 1024; j += 32) {
        v16bf b00 = load_frag(kb + (size_t)(j + r15) * 64 + half8);
        v16bf b01 = load_frag(kb + (size_t)(j + r15) * 64 + 32 + half8);
        v16bf b10 = load_frag(kb + (size_t)(j + 16 + r15) * 64 + half8);
        v16bf b11 = load_frag(kb + (size_t)(j + 16 + r15) * 64 + 32 + half8);
        v8f s0 = {}, s1 = {};
        s0 = wmma_bf16(qf0, b00, s0); s0 = wmma_bf16(qf1, b01, s0);
        s1 = wmma_bf16(qf0, b10, s1); s1 = wmma_bf16(qf1, b11, s1);

#pragma unroll
        for (int r = 0; r < 8; ++r) {
            float x0 = s0[r] * 0.125f, x1 = s1[r] * 0.125f;
            float cm = fmaxf(x0, x1);
#pragma unroll
            for (int msk = 8; msk; msk >>= 1) cm = fmaxf(cm, __shfl_xor(cm, msk, 32));
            float mn = fmaxf(mrow[r], cm);
            float alpha = __expf(mrow[r] - mn);
            float p0 = __expf(x0 - mn), p1 = __expf(x1 - mn);
            float rsum = p0 + p1;
#pragma unroll
            for (int msk = 8; msk; msk >>= 1) rsum += __shfl_xor(rsum, msk, 32);
            lrow[r] = lrow[r] * alpha + rsum;
            mrow[r] = mn;
            o0[r] *= alpha; o1[r] *= alpha; o2[r] *= alpha; o3[r] *= alpha;
            int mr = r + (halfsel << 3);
            pbuf[mr * 32 + r15] = (__bf16)p0;
            pbuf[mr * 32 + 16 + r15] = (__bf16)p1;
        }
        asm volatile("s_wait_dscnt 0" ::: "memory");
        v16bf pf = load_frag(&pbuf[r15 * 32 + half8]);
        v16bf vf0 = load_frag(vb + (size_t)(0 + r15) * 1024 + j + half8);
        v16bf vf1 = load_frag(vb + (size_t)(16 + r15) * 1024 + j + half8);
        v16bf vf2 = load_frag(vb + (size_t)(32 + r15) * 1024 + j + half8);
        v16bf vf3 = load_frag(vb + (size_t)(48 + r15) * 1024 + j + half8);
        o0 = wmma_bf16(pf, vf0, o0);
        o1 = wmma_bf16(pf, vf1, o1);
        o2 = wmma_bf16(pf, vf2, o2);
        o3 = wmma_bf16(pf, vf3, o3);
    }

    int b = bh >> 3, hh = bh & 7;
#pragma unroll
    for (int r = 0; r < 8; ++r) {
        int mr = r + (halfsel << 3);
        int token = qt * 16 + mr;
        float inv = 1.f / lrow[r];
        size_t base = ((size_t)(b * 1024 + token)) * 512 + hh * 64;
        o[base + 0  + r15] = (__bf16)(o0[r] * inv);
        o[base + 16 + r15] = (__bf16)(o1[r] * inv);
        o[base + 32 + r15] = (__bf16)(o2[r] * inv);
        o[base + 48 + r15] = (__bf16)(o3[r] * inv);
    }
}

// ---------------------------------------------------------------------------
// SwiGLU: ff[m,j] = silu(x12[m,j]) * x12[m,2048+j]   (2048-token chunk)
// ---------------------------------------------------------------------------
__global__ __launch_bounds__(256) void k_swiglu(const float* __restrict__ x12,
                                                __bf16* __restrict__ ff) {
    int i = blockIdx.x * 256 + threadIdx.x;  // 2048*2048
    int mloc = i >> 11, j = i & 2047;
    float x1 = x12[(size_t)mloc * 4096 + j];
    float x2 = x12[(size_t)mloc * 4096 + 2048 + j];
    ff[(size_t)mloc * 2048 + j] = (__bf16)((x1 / (1.f + __expf(-x1))) * x2);
}

// ---------------------------------------------------------------------------
// Final: out(B,C,H,W) = x + h(B,N,C)^T
// ---------------------------------------------------------------------------
__global__ __launch_bounds__(256) void k_final(const float* __restrict__ x,
                                               const float* __restrict__ h,
                                               float* __restrict__ out) {
    size_t i = (size_t)blockIdx.x * 256 + threadIdx.x;  // 16*512*1024
    int n = (int)(i & 1023);
    int c = (int)((i >> 10) & 511);
    int b = (int)(i >> 19);
    out[i] = x[i] + h[((size_t)b * 1024 + n) * 512 + c];
}

// ---------------------------------------------------------------------------
// Host orchestration
// ---------------------------------------------------------------------------
extern "C" void kernel_launch(void* const* d_in, const int* in_sizes, int n_in,
                              void* d_out, int out_size, void* d_ws, size_t ws_size,
                              hipStream_t stream) {
    const float* x      = (const float*)d_in[0];
    const float* temb   = (const float*)d_in[1];
    const float* gn_w   = (const float*)d_in[2];
    const float* gn_b   = (const float*)d_in[3];
    const float* ada_w  = (const float*)d_in[4];
    const float* ada_b  = (const float*)d_in[5];
    const float* n1_w   = (const float*)d_in[6];
    const float* n2_w   = (const float*)d_in[7];
    const float* qkv_w  = (const float*)d_in[8];
    const float* qkv_b  = (const float*)d_in[9];
    const float* qn_w   = (const float*)d_in[10];
    const float* kn_w   = (const float*)d_in[11];
    const float* proj_w = (const float*)d_in[12];
    const float* proj_b = (const float*)d_in[13];
    const float* w12    = (const float*)d_in[14];
    const float* b12    = (const float*)d_in[15];
    const float* w_out  = (const float*)d_in[16];
    const float* b_out  = (const float*)d_in[17];

    char* ws = (char*)d_ws;
    size_t off = 0;
    auto alloc = [&](size_t bytes) -> void* {
        void* p = ws + off;
        off += (bytes + 255) & ~(size_t)255;
        return p;
    };
    __bf16* Wqkv  = (__bf16*)alloc((size_t)1536 * 512 * 2);
    __bf16* Wproj = (__bf16*)alloc((size_t)512 * 512 * 2);
    __bf16* W12b  = (__bf16*)alloc((size_t)4096 * 512 * 2);
    __bf16* Woutb = (__bf16*)alloc((size_t)512 * 2048 * 2);
    float*  ada   = (float*)alloc((size_t)16 * 3072 * 4);
    float*  h     = (float*)alloc((size_t)16384 * 512 * 4);
    __bf16* hn    = (__bf16*)alloc((size_t)16384 * 512 * 2);
    __bf16* qb    = (__bf16*)alloc((size_t)16384 * 512 * 2);
    __bf16* kb    = (__bf16*)alloc((size_t)16384 * 512 * 2);
    __bf16* vT    = (__bf16*)alloc((size_t)16384 * 512 * 2);
    __bf16* ob    = (__bf16*)alloc((size_t)16384 * 512 * 2);
    float*  scr   = (float*)alloc((size_t)2048 * 4096 * 4);  // qkv-chunk / x12-chunk
    __bf16* ff    = (__bf16*)alloc((size_t)2048 * 2048 * 2);
    (void)ws_size; (void)in_sizes; (void)n_in; (void)out_size;

    // 1. weights -> bf16
    k_tobf16<<<(1536 * 512 + 255) / 256, 256, 0, stream>>>(qkv_w, Wqkv, 1536 * 512);
    k_tobf16<<<(512 * 512 + 255) / 256, 256, 0, stream>>>(proj_w, Wproj, 512 * 512);
    k_tobf16<<<(4096 * 512 + 255) / 256, 256, 0, stream>>>(w12, W12b, 4096 * 512);
    k_tobf16<<<(512 * 2048 + 255) / 256, 256, 0, stream>>>(w_out, Woutb, 512 * 2048);

    // 2. adaLN + GroupNorm
    k_adaln<<<16, 256, 0, stream>>>(temb, ada_w, ada_b, ada);
    k_gnorm<<<512, 256, 0, stream>>>(x, gn_w, gn_b, h);

    // 3. rmsnorm1 + modulate (shift=sm@0, scale=scm@512)
    k_rmsmod<<<16384, 256, 0, stream>>>(h, n1_w, ada, 0, 512, hn);

    // 4. chunked QKV GEMM + rmsnorm/rope
    for (int c = 0; c < 8; ++c) {
        int m0 = c * 2048;
        k_gemm<<<32 * 24, 32, 0, stream>>>(hn + (size_t)m0 * 512, Wqkv, qkv_b,
                                           scr, nullptr, nullptr,
                                           512, 1536, 24, 0, 0, 0);
        k_ropeqkv<<<2048, 256, 0, stream>>>(scr, qn_w, kn_w, qb, kb, vT, m0);
    }

    // 5. flash attention: 64 q-tiles x 128 (b,h)
    k_fattn<<<dim3(64, 128), 32, 0, stream>>>(qb, kb, vT, ob);

    // 6. proj GEMM, fused residual h += gm * (o + b)   (gm@1024)
    k_gemm<<<256 * 8, 32, 0, stream>>>(ob, Wproj, proj_b, nullptr, h, ada,
                                       512, 512, 8, 0, 1024, 1);

    // 7. rmsnorm2 + modulate (shift=sp@1536, scale=scp@2048)
    k_rmsmod<<<16384, 256, 0, stream>>>(h, n2_w, ada, 1536, 2048, hn);

    // 8. chunked SwiGLU FFN, fused residual h += gp * (...)   (gp@2560)
    for (int c = 0; c < 8; ++c) {
        int m0 = c * 2048;
        k_gemm<<<32 * 64, 32, 0, stream>>>(hn + (size_t)m0 * 512, W12b, b12,
                                           scr, nullptr, nullptr,
                                           512, 4096, 64, 0, 0, 0);
        k_swiglu<<<(2048 * 2048) / 256, 256, 0, stream>>>(scr, ff);
        k_gemm<<<32 * 8, 32, 0, stream>>>(ff, Woutb, b_out, nullptr, h, ada,
                                          2048, 512, 8, m0, 2560, 1);
    }

    // 9. out = x + h^T
    k_final<<<(16 * 512 * 1024) / 256, 256, 0, stream>>>(x, h, (float*)d_out);
}
